// SelfAttention_35622458753629
// MI455X (gfx1250) — matrix operand order repaired and checked by reading, not compiled, not run on hardware
//
#include <hip/hip_runtime.h>
#include <hip/hip_bf16.h>

typedef __attribute__((ext_vector_type(16))) _Float16 v16h;
typedef __attribute__((ext_vector_type(8)))  _Float16 v8h;
typedef __attribute__((ext_vector_type(8)))  float    v8f;
typedef __attribute__((ext_vector_type(4)))  int      v4i;

#define DIM      1024
#define SEQ      1365
#define SP       1376          // SEQ padded to multiple of 16
#define NB       8
#define NH       16
#define HD       64
#define MROWS    (NB*SEQ)      // 10920
#define MPAD     10928         // padded to multiple of 16
#define TDIM     3072

#if defined(__has_builtin)
# if __has_builtin(__builtin_amdgcn_global_load_async_to_lds_b128) && \
     __has_builtin(__builtin_amdgcn_s_wait_asynccnt)
#  define USE_ASYNC_LDS 1
# endif
#endif

// ---- segment helpers (LENGTHS = 1,4,16,64,256,1024; starts 0,1,5,21,85,341) ----
__device__ __forceinline__ int seg_start_of(int k) {
    if (k < 1)    return 0;
    if (k < 5)    return 1;
    if (k < 21)   return 5;
    if (k < 85)   return 21;
    if (k < 341)  return 85;
    if (k < 1365) return 341;
    return 1 << 29;            // padded keys: never visible
}
__device__ __forceinline__ int seg_end_of(int q) {
    if (q < 1)   return 1;
    if (q < 5)   return 5;
    if (q < 21)  return 21;
    if (q < 85)  return 85;
    if (q < 341) return 341;
    return 1365;
}

// ================= adaLN: wb[b][n] = cond[b] @ adaln_w + adaln_b =================
__global__ void adaln_kernel(const float* __restrict__ cond,
                             const float* __restrict__ w,
                             const float* __restrict__ b,
                             float* __restrict__ wb) {
    int idx = blockIdx.x * blockDim.x + threadIdx.x;
    if (idx >= NB * 2 * DIM) return;
    int bb = idx >> 11, n = idx & 2047;
    const float* c = cond + bb * 256;
    float acc = b[n];
    for (int k = 0; k < 256; ++k) acc += c[k] * w[(size_t)k * 2048 + n];
    wb[idx] = acc;
}

// ============ transpose f32 [K][N] -> f16 [N][K] (K-contiguous for B-fragments) ============
__global__ void transpose_f32_to_f16(const float* __restrict__ in,
                                     _Float16* __restrict__ out, int K, int N) {
    int idx = blockIdx.x * blockDim.x + threadIdx.x;
    if (idx >= K * N) return;
    int n = idx / K, k = idx - n * K;
    out[idx] = (_Float16)in[(size_t)k * N + n];
}

// ================= LayerNorm + adaLN modulate -> f16 (zero the pad rows) =================
__global__ void ln_mod_kernel(const float* __restrict__ x,
                              const float* __restrict__ wb,
                              _Float16* __restrict__ xn) {
    int row = blockIdx.x, t = threadIdx.x;
    if (row >= MROWS) {
        for (int i = 0; i < 4; ++i) xn[(size_t)row * DIM + t + 256 * i] = (_Float16)0.0f;
        return;
    }
    int b = row / SEQ;
    const float* xr = x + (size_t)row * DIM;
    float v[4], s = 0.f, s2 = 0.f;
    for (int i = 0; i < 4; ++i) { v[i] = xr[t + 256 * i]; s += v[i]; s2 += v[i] * v[i]; }
    __shared__ float rs[256], rq[256];
    rs[t] = s; rq[t] = s2; __syncthreads();
    for (int off = 128; off; off >>= 1) {
        if (t < off) { rs[t] += rs[t + off]; rq[t] += rq[t + off]; }
        __syncthreads();
    }
    float mu  = rs[0] * (1.0f / DIM);
    float var = rq[0] * (1.0f / DIM) - mu * mu;
    float inv = rsqrtf(var + 1e-5f);
    for (int i = 0; i < 4; ++i) {
        int c = t + 256 * i;
        float w1 = wb[b * 2048 + c] + 1.0f;
        float b1 = wb[b * 2048 + 1024 + c];
        xn[(size_t)row * DIM + c] = (_Float16)(((v[i] - mu) * inv) * w1 + b1);
    }
}

// ---------------- WMMA fragment helpers ----------------
__device__ __forceinline__ v16h make_a(v8h lo, v8h hi) {
    v16h a;
    #pragma unroll
    for (int i = 0; i < 8; ++i) { a[i] = lo[i]; a[8 + i] = hi[i]; }
    return a;
}
__device__ __forceinline__ v8f zero8() {
    v8f z;
    #pragma unroll
    for (int i = 0; i < 8; ++i) z[i] = 0.0f;
    return z;
}

// ================= QKV GEMM: block-cooperative 128x64 macro-tile =================
// 8 waves/block, each wave owns 16 rows x 64 cols (4 accumulators).
// B panel (64 n-rows x 32 k) staged into LDS via async loads, double-buffered.
// Epilogue scatters into q16/k16 [b,h,s,d] and vT16 [b,h,d,s] (f16) with bias.
__global__ __launch_bounds__(256) void qkv_gemm(const _Float16* __restrict__ xn,
                                                const _Float16* __restrict__ wT,
                                                const float* __restrict__ bias,
                                                _Float16* __restrict__ q16,
                                                _Float16* __restrict__ k16,
                                                _Float16* __restrict__ vT16) {
    int wave = threadIdx.x >> 5, lane = threadIdx.x & 31;
    int lm = lane & 15, h4 = lane >> 4;
    int n0 = blockIdx.x * 64;                    // 48 groups
    int m0 = blockIdx.y * 128 + wave * 16;       // 86 groups (rows clamped for loads)

    int arowi = m0 + lm; if (arowi > MPAD - 1) arowi = MPAD - 1;
    const _Float16* arow = xn + (size_t)arowi * DIM;

    v8f acc[4];
    #pragma unroll
    for (int t = 0; t < 4; ++t) acc[t] = zero8();

#ifdef USE_ASYNC_LDS
    __shared__ __attribute__((aligned(32))) _Float16 bstage[2][64 * 32];
    // cooperative stage of 64x32 f16 B panel: one async b128 per lane (256 lanes = 4KB)
    auto stage = [&](int k0, int buf) {
        int nl = threadIdx.x >> 2, kc = threadIdx.x & 3;
        const _Float16* src = wT + (size_t)(n0 + nl) * DIM + k0 + kc * 8;
        _Float16* dst = &bstage[buf][nl * 32 + kc * 8];
        __builtin_amdgcn_global_load_async_to_lds_b128(
            (__attribute__((address_space(1))) v4i*)(void*)src,
            (__attribute__((address_space(3))) v4i*)(void*)dst, 0, 0);
    };
    stage(0, 0);
    for (int k0 = 0; k0 < DIM; k0 += 32) {
        int cur = (k0 >> 5) & 1;
        if (k0 + 32 < DIM) { stage(k0 + 32, cur ^ 1); __builtin_amdgcn_s_wait_asynccnt(1); }
        else               { __builtin_amdgcn_s_wait_asynccnt(0); }
        __syncthreads();                         // panel visible to all waves
        v16h a = make_a(*(const v8h*)(arow + k0 + 16 * h4),
                        *(const v8h*)(arow + k0 + 16 * h4 + 8));
        const _Float16* bb = &bstage[cur][0];
        #pragma unroll
        for (int t = 0; t < 4; ++t) {
            v16h bf = *(const v16h*)(bb + (t * 16 + lm) * 32 + 16 * h4);
            acc[t] = __builtin_amdgcn_wmma_f32_16x16x32_f16(false, a, false, bf,
                                                            (short)0, acc[t], false, false);
        }
        __syncthreads();                         // reads done before buffer is re-staged
    }
#else
    for (int k0 = 0; k0 < DIM; k0 += 32) {
        v16h a = make_a(*(const v8h*)(arow + k0 + 16 * h4),
                        *(const v8h*)(arow + k0 + 16 * h4 + 8));
        #pragma unroll
        for (int t = 0; t < 4; ++t) {
            v16h bf = *(const v16h*)(wT + (size_t)(n0 + t * 16 + lm) * DIM + k0 + 16 * h4);
            acc[t] = __builtin_amdgcn_wmma_f32_16x16x32_f16(false, a, false, bf,
                                                            (short)0, acc[t], false, false);
        }
    }
#endif

    #pragma unroll
    for (int t = 0; t < 4; ++t) {
        int n = n0 + t * 16 + lm;
        float bn = bias[n];
        int which = n >> 10, hh = (n >> 6) & 15, d = n & 63;
        #pragma unroll
        for (int r = 0; r < 8; ++r) {
            int row = m0 + r + 8 * h4;
            if (row < MROWS) {
                int b = row / SEQ, s = row - b * SEQ;
                _Float16 val = (_Float16)(acc[t][r] + bn);
                size_t bh = (size_t)(b * NH + hh);
                if (which == 0)      q16[(bh * SP + s) * HD + d] = val;
                else if (which == 1) k16[(bh * SP + s) * HD + d] = val;
                else                 vT16[(bh * HD + d) * SP + s] = val;
            }
        }
    }
}

// ================= RoPE on q,k (in place, f16); theta recomputed on device =================
__global__ void rope_kernel(_Float16* __restrict__ q16, _Float16* __restrict__ k16) {
    int tid = blockIdx.x * blockDim.x + threadIdx.x;
    if (tid >= NB * NH * SEQ) return;
    int s = tid % SEQ, bh = tid / SEQ, h = bh & 15;
    float ph = 0.f, pw = 0.f, pr = 0.f;
    if (s > 0) {
        int start, side; float prv;
        if (s < 5)        { start = 1;   side = 2;  prv = 0.2f; }
        else if (s < 21)  { start = 5;   side = 4;  prv = 0.4f; }
        else if (s < 85)  { start = 21;  side = 8;  prv = 0.6f; }
        else if (s < 341) { start = 85;  side = 16; prv = 0.8f; }
        else              { start = 341; side = 32; prv = 1.0f; }
        int t = s - start, rr = t / side, cc = t - rr * side;
        float inv = 1.0f / (float)side;
        ph = -1.0f + (float)(2 * rr + 1) * inv;
        pw = -1.0f + (float)(2 * cc + 1) * inv;
        pr = prv;
    }
    const float lmin = 1.14472988584940017f;                 // ln(pi)
    const float step = (3.44731497884344704f - 1.14472988584940017f) * (1.0f / 128.0f);
    _Float16* qr = q16 + ((size_t)bh * SP + s) * HD;
    _Float16* kr = k16 + ((size_t)bh * SP + s) * HD;
    for (int kk = 0; kk < 24; ++kk) {
        int j = kk & 7, g = kk >> 3;
        float comp = (g == 0) ? ph : ((g == 1) ? pw : pr);
        float fr = expf(lmin + (float)(j * 16 + h) * step);
        float th = comp * fr;
        float c = cosf(th), sn = sinf(th);
        float a1 = (float)qr[kk], a2 = (float)qr[kk + 24];
        qr[kk]      = (_Float16)(a1 * c - a2 * sn);
        qr[kk + 24] = (_Float16)(a2 * c + a1 * sn);
        float b1 = (float)kr[kk], b2 = (float)kr[kk + 24];
        kr[kk]      = (_Float16)(b1 * c - b2 * sn);
        kr[kk + 24] = (_Float16)(b2 * c + b1 * sn);
    }
}

// ================= Flash attention: one wave per (b,h, 16-query tile) =================
__global__ __launch_bounds__(128) void attn_kernel(const _Float16* __restrict__ q16,
                                                   const _Float16* __restrict__ k16,
                                                   const _Float16* __restrict__ vT16,
                                                   _Float16* __restrict__ o16) {
    __shared__ __attribute__((aligned(32))) _Float16 plds[4][16 * 32];
    const int QT = SP / 16;                                  // 86
    int wave = threadIdx.x >> 5, lane = threadIdx.x & 31;
    int lm = lane & 15, h4 = lane >> 4;
    int tile = blockIdx.x * 4 + wave;
    if (tile >= NB * NH * QT) return;
    int qt = tile % QT, bh = tile / QT;
    int q0 = qt * 16;

    int qrow = q0 + lm; if (qrow > SEQ - 1) qrow = SEQ - 1;
    const _Float16* qb = q16 + ((size_t)bh * SP + qrow) * HD;
    v16h qa0 = make_a(*(const v8h*)(qb + 16 * h4),      *(const v8h*)(qb + 16 * h4 + 8));
    v16h qa1 = make_a(*(const v8h*)(qb + 32 + 16 * h4), *(const v8h*)(qb + 40 + 16 * h4));

    float m[8], l[8];
    v8f oacc[4];
    #pragma unroll
    for (int r = 0; r < 8; ++r) { m[r] = -1e30f; l[r] = 0.0f; }
    #pragma unroll
    for (int t = 0; t < 4; ++t) oacc[t] = zero8();

    int qmax = q0 + 15; if (qmax > SEQ - 1) qmax = SEQ - 1;
    int kend = seg_end_of(qmax);
    const _Float16* kbase = k16 + (size_t)bh * SP * HD;
    const _Float16* vbase = vT16 + (size_t)bh * HD * SP;
    _Float16* pl = plds[wave];

    for (int j0 = 0; j0 < kend; j0 += 32) {
        int key0 = j0 + lm, key1 = j0 + 16 + lm;
        v16h b00 = *(const v16h*)(kbase + (size_t)key0 * HD + 16 * h4);
        v16h b01 = *(const v16h*)(kbase + (size_t)key0 * HD + 32 + 16 * h4);
        v16h b10 = *(const v16h*)(kbase + (size_t)key1 * HD + 16 * h4);
        v16h b11 = *(const v16h*)(kbase + (size_t)key1 * HD + 32 + 16 * h4);
        v8f s0 = zero8(), s1 = zero8();
        s0 = __builtin_amdgcn_wmma_f32_16x16x32_f16(false, qa0, false, b00, (short)0, s0, false, false);
        s0 = __builtin_amdgcn_wmma_f32_16x16x32_f16(false, qa1, false, b01, (short)0, s0, false, false);
        s1 = __builtin_amdgcn_wmma_f32_16x16x32_f16(false, qa0, false, b10, (short)0, s1, false, false);
        s1 = __builtin_amdgcn_wmma_f32_16x16x32_f16(false, qa1, false, b11, (short)0, s1, false, false);

        int ss0 = seg_start_of(key0), ss1 = seg_start_of(key1);
        #pragma unroll
        for (int r = 0; r < 8; ++r) {
            int q = q0 + r + 8 * h4;
            float x0 = (ss0 <= q) ? s0[r] * 0.125f : -1e30f;
            float x1 = (ss1 <= q) ? s1[r] * 0.125f : -1e30f;
            float mx = fmaxf(x0, x1);
            for (int off = 8; off; off >>= 1) mx = fmaxf(mx, __shfl_xor(mx, off, 32));
            float mn = fmaxf(m[r], mx);
            float alpha = expf(m[r] - mn);
            float p0 = expf(x0 - mn), p1 = expf(x1 - mn);
            float rsum = p0 + p1;
            for (int off = 8; off; off >>= 1) rsum += __shfl_xor(rsum, off, 32);
            l[r] = l[r] * alpha + rsum;
            m[r] = mn;
            oacc[0][r] *= alpha; oacc[1][r] *= alpha;
            oacc[2][r] *= alpha; oacc[3][r] *= alpha;
            pl[(r + 8 * h4) * 32 + lm]      = (_Float16)p0;
            pl[(r + 8 * h4) * 32 + 16 + lm] = (_Float16)p1;
        }
        __builtin_amdgcn_wave_barrier();   // keep LDS store->load order for cross-lane P exchange
        v16h pa = make_a(*(const v8h*)(pl + lm * 32 + 16 * h4),
                         *(const v8h*)(pl + lm * 32 + 16 * h4 + 8));
        #pragma unroll
        for (int t = 0; t < 4; ++t) {
            v16h bv = *(const v16h*)(vbase + (size_t)(t * 16 + lm) * SP + j0 + 16 * h4);
            oacc[t] = __builtin_amdgcn_wmma_f32_16x16x32_f16(false, pa, false, bv,
                                                             (short)0, oacc[t], false, false);
        }
        __builtin_amdgcn_wave_barrier();
    }

    int b = bh >> 4, hh = bh & 15;
    #pragma unroll
    for (int r = 0; r < 8; ++r) {
        int q = q0 + r + 8 * h4;
        if (q < SEQ) {
            float inv = 1.0f / l[r];
            size_t rowbase = ((size_t)b * SEQ + q) * DIM + hh * HD;
            #pragma unroll
            for (int t = 0; t < 4; ++t)
                o16[rowbase + t * 16 + lm] = (_Float16)(oacc[t][r] * inv);
        }
    }
}

// ================= Out projection + bias + residual (16x64 per wave) =================
__global__ __launch_bounds__(256) void out_gemm(const _Float16* __restrict__ o16,
                                                const _Float16* __restrict__ wT,
                                                const float* __restrict__ bias,
                                                const float* __restrict__ x,
                                                float* __restrict__ out) {
    const int NG = DIM / 64;                         // 16 n-groups of 64
    int wave = threadIdx.x >> 5, lane = threadIdx.x & 31;
    int tile = blockIdx.x * 8 + wave;
    int mt = tile / NG, ng = tile - mt * NG;
    int n0 = ng * 64;
    int lm = lane & 15, h4 = lane >> 4;
    const _Float16* arow = o16 + (size_t)(mt * 16 + lm) * DIM;
    v8f acc[4];
    #pragma unroll
    for (int t = 0; t < 4; ++t) acc[t] = zero8();
    for (int k0 = 0; k0 < DIM; k0 += 32) {
        v16h a = make_a(*(const v8h*)(arow + k0 + 16 * h4),
                        *(const v8h*)(arow + k0 + 16 * h4 + 8));
        #pragma unroll
        for (int t = 0; t < 4; ++t) {
            v16h bf = *(const v16h*)(wT + (size_t)(n0 + t * 16 + lm) * DIM + k0 + 16 * h4);
            acc[t] = __builtin_amdgcn_wmma_f32_16x16x32_f16(false, a, false, bf,
                                                            (short)0, acc[t], false, false);
        }
    }
    #pragma unroll
    for (int t = 0; t < 4; ++t) {
        int n = n0 + t * 16 + lm;
        float bn = bias[n];
        #pragma unroll
        for (int r = 0; r < 8; ++r) {
            int row = mt * 16 + r + 8 * h4;
            if (row < MROWS)
                out[(size_t)row * DIM + n] = acc[t][r] + bn + x[(size_t)row * DIM + n];
        }
    }
}

extern "C" void kernel_launch(void* const* d_in, const int* in_sizes, int n_in,
                              void* d_out, int out_size, void* d_ws, size_t ws_size,
                              hipStream_t stream) {
    const float* x        = (const float*)d_in[0];
    const float* cond     = (const float*)d_in[1];
    const float* adaln_w  = (const float*)d_in[2];
    const float* adaln_b  = (const float*)d_in[3];
    const float* qkv_w    = (const float*)d_in[4];
    const float* qkv_b    = (const float*)d_in[5];
    const float* out_w    = (const float*)d_in[6];
    const float* out_b    = (const float*)d_in[7];
    float* out = (float*)d_out;

    char* p = (char*)d_ws;
    auto carve = [&](size_t bytes) -> char* {
        char* r = p; p += (bytes + 255) & ~(size_t)255; return r;
    };
    float*    wb    = (float*)   carve((size_t)NB * 2 * DIM * 4);
    _Float16* xn16  = (_Float16*)carve((size_t)MPAD * DIM * 2);      // reused as o16
    _Float16* qkvT  = (_Float16*)carve((size_t)TDIM * DIM * 2);
    _Float16* outT  = (_Float16*)carve((size_t)DIM * DIM * 2);
    _Float16* q16   = (_Float16*)carve((size_t)NB * NH * SP * HD * 2);
    _Float16* k16   = (_Float16*)carve((size_t)NB * NH * SP * HD * 2);
    _Float16* vT16  = (_Float16*)carve((size_t)NB * NH * SP * HD * 2);

    adaln_kernel<<<(NB * 2 * DIM + 255) / 256, 256, 0, stream>>>(cond, adaln_w, adaln_b, wb);
    transpose_f32_to_f16<<<(DIM * TDIM + 255) / 256, 256, 0, stream>>>(qkv_w, qkvT, DIM, TDIM);
    transpose_f32_to_f16<<<(DIM * DIM + 255) / 256, 256, 0, stream>>>(out_w, outT, DIM, DIM);
    ln_mod_kernel<<<MPAD, 256, 0, stream>>>(x, wb, xn16);
    qkv_gemm<<<dim3(TDIM / 64, (MPAD + 127) / 128), 256, 0, stream>>>(xn16, qkvT, qkv_b,
                                                                      q16, k16, vT16);
    rope_kernel<<<(NB * NH * SEQ + 255) / 256, 256, 0, stream>>>(q16, k16);
    attn_kernel<<<NB * NH * (SP / 16) / 4, 128, 0, stream>>>(q16, k16, vT16, xn16);
    out_gemm<<<(MPAD / 16) * (DIM / 64) / 8, 256, 0, stream>>>(xn16, outT, out_b, x, out);
}
